// MultiHeadAttention_16458314678882
// MI455X (gfx1250) — matrix operand order repaired and checked by reading, not compiled, not run on hardware
//
#include <hip/hip_runtime.h>
#include <hip/hip_bf16.h>

#define BB 4
#define SS 2048
#define DD 1024
#define HH 16
#define HD 64

typedef __bf16 bf16_t;
typedef __attribute__((ext_vector_type(16))) __bf16 v16bf;
typedef __attribute__((ext_vector_type(8)))  __bf16 v8bf;
typedef __attribute__((ext_vector_type(8)))  float   v8f;

#if defined(__HIP_DEVICE_COMPILE__) && __has_builtin(__builtin_amdgcn_global_load_async_to_lds_b128)
#define HAVE_ASYNC 1
#else
#define HAVE_ASYNC 0
#endif

// ---------- helpers ----------

__device__ __forceinline__ bf16_t f2bf(float f) {
  unsigned u = __builtin_bit_cast(unsigned, f);
  unsigned r = (u + 0x7FFFu + ((u >> 16) & 1u)) >> 16;  // round-to-nearest-even
  unsigned short s = (unsigned short)r;
  return __builtin_bit_cast(bf16_t, s);
}

__device__ __forceinline__ v8f vzero8() {
  v8f z;
#pragma unroll
  for (int i = 0; i < 8; ++i) z[i] = 0.0f;
  return z;
}

// A-fragment, 16x32 bf16, from row-major storage (row stride = ld elements).
// Lane l: m = l&15; halves 0..7 -> k = 8g..8g+7, halves 8..15 -> k = 16+8g..
__device__ __forceinline__ v16bf load_a32(const bf16_t* base, int ld) {
  const int lane = threadIdx.x & 31;
  const bf16_t* p = base + (size_t)(lane & 15) * ld + ((lane >> 4) << 3);
  v8bf lo = *(const v8bf*)(p);
  v8bf hi = *(const v8bf*)(p + 16);
  v16bf r;
#pragma unroll
  for (int i = 0; i < 8; ++i) { r[i] = lo[i]; r[8 + i] = hi[i]; }
  return r;
}

// B-fragment, 32x16 bf16, from n-major storage (column n stride = ld elements).
// Lane l: n = l&15; halves 0..7 -> k = 8g..8g+7, halves 8..15 -> k = 16+8g..
__device__ __forceinline__ v16bf load_b32(const bf16_t* base, int ld) {
  const int lane = threadIdx.x & 31;
  const bf16_t* p = base + (size_t)(lane & 15) * ld + ((lane >> 4) << 3);
  v8bf lo = *(const v8bf*)(p);
  v8bf hi = *(const v8bf*)(p + 16);
  v16bf r;
#pragma unroll
  for (int i = 0; i < 8; ++i) { r[i] = lo[i]; r[8 + i] = hi[i]; }
  return r;
}

__device__ __forceinline__ v8f wmma_bf16(v16bf a, v16bf b, v8f c) {
  return __builtin_amdgcn_wmma_f32_16x16x32_bf16(
      /*neg_a=*/false, a, /*neg_b=*/false, b,
      /*c_mod=*/(short)0, c, /*reuse_a=*/false, /*reuse_b=*/false);
}

// ---------- async-to-LDS plumbing (gfx1250 ASYNCcnt path) ----------

#if HAVE_ASYNC
// Builtin prototype (from diagnostic): param0 = generic `int __vector(4)*`,
// param1 = LDS (addrspace 3) `int __vector(4)*`, then imm offset, imm cpol.
typedef int v4i_gcc __attribute__((vector_size(16)));
typedef __attribute__((address_space(3))) v4i_gcc* lds_v4i_ptr;
#endif

// Copy 16 bytes from global to LDS. Async on gfx1250 (ASYNCcnt-tracked DMA
// into LDS); falls back to load+ds_store if the builtin is unavailable.
__device__ __forceinline__ void async_copy16(const bf16_t* g, bf16_t* l) {
#if HAVE_ASYNC
  __builtin_amdgcn_global_load_async_to_lds_b128(
      (v4i_gcc*)g,
      (lds_v4i_ptr)(unsigned)(unsigned long long)l,
      0, 0);
#else
  *(v8bf*)l = *(const v8bf*)g;
#endif
}

template <int N>
__device__ __forceinline__ void wait_async() {
#if HAVE_ASYNC
#if __has_builtin(__builtin_amdgcn_s_wait_asynccnt)
  __builtin_amdgcn_s_wait_asynccnt(N);
#else
  asm volatile("s_wait_asynccnt %0" ::"i"(N) : "memory");
#endif
#endif
}

// ---------- prep kernels ----------

__global__ void k_cvt_x(const float* __restrict__ x, bf16_t* __restrict__ xb, int n) {
  int i = (blockIdx.x * blockDim.x + threadIdx.x) << 2;
  if (i + 3 < n) {
    float4 f = *(const float4*)(x + i);
    xb[i + 0] = f2bf(f.x);
    xb[i + 1] = f2bf(f.y);
    xb[i + 2] = f2bf(f.z);
    xb[i + 3] = f2bf(f.w);
  }
}

// w[Hn][Dn][En] (f32) -> wt[Hn][En][Dn] (bf16)
__global__ void k_transpose(const float* __restrict__ w, bf16_t* __restrict__ wt,
                            int Hn, int Dn, int En) {
  int idx = blockIdx.x * blockDim.x + threadIdx.x;
  int total = Hn * Dn * En;
  if (idx >= total) return;
  int d = idx % Dn;
  int t = idx / Dn;
  int e = t % En;
  int h = t / En;
  wt[idx] = f2bf(w[((size_t)h * Dn + d) * En + e]);
}

// ---------- QKV projection: Q,K row-major [B,H,S,HD]; V transposed [B,H,HD,S] ----------

__global__ __launch_bounds__(256) void k_qkv(
    const bf16_t* __restrict__ xb, const bf16_t* __restrict__ wqt,
    const bf16_t* __restrict__ wkt, const bf16_t* __restrict__ wvt,
    const float* __restrict__ bq, const float* __restrict__ bk,
    const float* __restrict__ bv,
    bf16_t* __restrict__ Qb, bf16_t* __restrict__ Kb, bf16_t* __restrict__ VTb) {
  const int wg = blockIdx.x * 8 + (threadIdx.x >> 5);
  const int mt = wg & 511;          // 512 row tiles of 16 over B*S
  const int h  = wg >> 9;           // 16 heads
  const int row0 = mt << 4;
  const int b  = row0 >> 11;        // S = 2048
  const int s0 = row0 & (SS - 1);
  const int lane = threadIdx.x & 31;
  const int n = lane & 15, g = lane >> 4;

  v8f aq[4], ak[4], av[4];
#pragma unroll
  for (int t = 0; t < 4; ++t) { aq[t] = vzero8(); ak[t] = vzero8(); av[t] = vzero8(); }

  for (int kk = 0; kk < DD; kk += 32) {
    v16bf a = load_a32(xb + (size_t)row0 * DD + kk, DD);
#pragma unroll
    for (int t = 0; t < 4; ++t) {
      v16bf bwq = load_b32(wqt + ((size_t)(h * HD + t * 16)) * DD + kk, DD);
      aq[t] = wmma_bf16(a, bwq, aq[t]);
      v16bf bwk = load_b32(wkt + ((size_t)(h * HD + t * 16)) * DD + kk, DD);
      ak[t] = wmma_bf16(a, bwk, ak[t]);
      v16bf bwv = load_b32(wvt + ((size_t)(h * HD + t * 16)) * DD + kk, DD);
      av[t] = wmma_bf16(a, bwv, av[t]);
    }
  }

  bf16_t* Qp  = Qb  + (size_t)(b * HH + h) * SS * HD;
  bf16_t* Kp  = Kb  + (size_t)(b * HH + h) * SS * HD;
  bf16_t* VTp = VTb + (size_t)(b * HH + h) * HD * SS;
  const float qscale = 0.125f;  // 1/sqrt(HD), folded into Q

#pragma unroll
  for (int t = 0; t < 4; ++t) {
    int e = t * 16 + n;
    float biq = bq[h * HD + e];
    float bik = bk[h * HD + e];
    float biv = bv[h * HD + e];
#pragma unroll
    for (int r = 0; r < 8; ++r) {
      int m = r + (g << 3);
      Qp[(size_t)(s0 + m) * HD + e] = f2bf((aq[t][r] + biq) * qscale);
      Kp[(size_t)(s0 + m) * HD + e] = f2bf(ak[t][r] + bik);
      VTp[(size_t)e * SS + (s0 + m)] = f2bf(av[t][r] + biv);
    }
  }
}

// ---------- flash attention ----------
// Block = 8 waves = 8 consecutive 16-query tiles of one (b,h).
// K/V 32-key tiles staged in LDS via double-buffered async-to-LDS copies,
// shared by all 8 waves (8x reduction in K/V traffic vs per-wave streaming).

__global__ __launch_bounds__(256) void k_attn(
    const bf16_t* __restrict__ Qb, const bf16_t* __restrict__ Kb,
    const bf16_t* __restrict__ VTb, bf16_t* __restrict__ Ocat) {
  __shared__ bf16_t kbuf[2][32 * HD];   // keys kb..kb+31, row-major [32][64]  (4KB each)
  __shared__ bf16_t vbuf[2][HD * 32];   // V^T rows e=0..63 x 32 keys [64][32] (4KB each)
  __shared__ bf16_t pbuf[8][16 * 32];   // per-wave P tile (16x32)             (8KB)

  const int w   = threadIdx.x >> 5;
  const int tid = threadIdx.x;
  const int blk = blockIdx.x;            // 1024 blocks
  const int qblk = blk & 15;             // 16 blocks of 128 queries per (b,h)
  const int h    = (blk >> 4) & 15;
  const int b    = blk >> 8;
  const int q0   = qblk * 128 + w * 16;  // this wave's query tile
  const int lane = threadIdx.x & 31;
  const int n = lane & 15, g = lane >> 4;

  const bf16_t* Qp  = Qb  + (size_t)(b * HH + h) * SS * HD;
  const bf16_t* Kp  = Kb  + (size_t)(b * HH + h) * SS * HD;
  const bf16_t* VTp = VTb + (size_t)(b * HH + h) * HD * SS;

  v16bf a0 = load_a32(Qp + (size_t)q0 * HD + 0, HD);
  v16bf a1 = load_a32(Qp + (size_t)q0 * HD + 32, HD);

  v8f acc[4];
  float mrow[8], lrow[8];
#pragma unroll
  for (int t = 0; t < 4; ++t) acc[t] = vzero8();
#pragma unroll
  for (int r = 0; r < 8; ++r) { mrow[r] = -3.0e38f; lrow[r] = 0.0f; }

  const int kend_wave = q0 + 16;         // causal: keys 0 .. q0+15
  const int nb = qblk * 4 + 4;           // key blocks for whole query block

  // Cooperative tile fetch: K tile is one contiguous 4KB span (16B/thread);
  // V^T tile is 64 rows x 64B (each thread: one 16B segment).
  const int vrow = tid >> 2, vseg = tid & 3;

  // preload block 0 into buffer 0
  async_copy16(Kp + /*kb=0*/ (size_t)0 * HD + tid * 8, &kbuf[0][tid * 8]);
  async_copy16(VTp + (size_t)vrow * SS + 0 + vseg * 8, &vbuf[0][vrow * 32 + vseg * 8]);

  for (int ib = 0; ib < nb; ++ib) {
    const int kb  = ib * 32;
    const int cur = ib & 1;
    if (ib + 1 < nb) {
      const int kn = kb + 32;
      async_copy16(Kp + (size_t)kn * HD + tid * 8, &kbuf[cur ^ 1][tid * 8]);
      async_copy16(VTp + (size_t)vrow * SS + kn + vseg * 8,
                   &vbuf[cur ^ 1][vrow * 32 + vseg * 8]);
      wait_async<2>();   // in-order completion => current buffer resident
    } else {
      wait_async<0>();
    }
    __syncthreads();     // all waves' copies for `cur` are visible

    if (kb < kend_wave) {  // wave-uniform causal guard (EXEC stays all-ones)
      // scores: S[16q x 32k] from LDS K tile, reduction over HD=64
      v8f s0t = vzero8(), s1t = vzero8();
      v16bf b00 = load_b32(&kbuf[cur][0], HD);
      v16bf b01 = load_b32(&kbuf[cur][32], HD);
      s0t = wmma_bf16(a0, b00, s0t);
      s0t = wmma_bf16(a1, b01, s0t);
      v16bf b10 = load_b32(&kbuf[cur][16 * HD + 0], HD);
      v16bf b11 = load_b32(&kbuf[cur][16 * HD + 32], HD);
      s1t = wmma_bf16(a0, b10, s1t);
      s1t = wmma_bf16(a1, b11, s1t);

      if (kb + 31 > q0) {  // causal mask via lane selects
#pragma unroll
        for (int r = 0; r < 8; ++r) {
          int m = q0 + r + (g << 3);
          if (kb + n > m)      s0t[r] = -3.0e38f;
          if (kb + 16 + n > m) s1t[r] = -3.0e38f;
        }
      }

      // online softmax; row stats replicated across each 16-lane half
      float nm[8], sc[8];
#pragma unroll
      for (int r = 0; r < 8; ++r) {
        float v = fmaxf(s0t[r], s1t[r]);
#pragma unroll
        for (int msk = 1; msk < 16; msk <<= 1) v = fmaxf(v, __shfl_xor(v, msk, 32));
        nm[r] = fmaxf(mrow[r], v);
        sc[r] = __expf(mrow[r] - nm[r]);
      }
#pragma unroll
      for (int r = 0; r < 8; ++r) {
        float p0 = __expf(s0t[r] - nm[r]);
        float p1 = __expf(s1t[r] - nm[r]);
        int m = r + (g << 3);
        pbuf[w][m * 32 + n]      = f2bf(p0);  // C-layout -> row-major in LDS
        pbuf[w][m * 32 + 16 + n] = f2bf(p1);
        float rs = p0 + p1;
#pragma unroll
        for (int msk = 1; msk < 16; msk <<= 1) rs += __shfl_xor(rs, msk, 32);
        lrow[r] = lrow[r] * sc[r] + rs;
        mrow[r] = nm[r];
#pragma unroll
        for (int t = 0; t < 4; ++t) acc[t][r] *= sc[r];
      }

      // P back as A-fragment (same-wave LDS ops are in-order; no barrier needed)
      v16bf ap = load_a32(&pbuf[w][0], 32);
#pragma unroll
      for (int t = 0; t < 4; ++t) {
        v16bf bvf = load_b32(&vbuf[cur][(t * 16) * 32], 32);
        acc[t] = wmma_bf16(ap, bvf, acc[t]);
      }
    }
    __syncthreads();     // protect `cur` buffer before it is refilled
  }

  // normalize + write concat-head output [B*S, D] bf16
#pragma unroll
  for (int r = 0; r < 8; ++r) {
    float inv = 1.0f / lrow[r];
    int m = r + (g << 3);
    size_t rowbase = ((size_t)b * SS + q0 + m) * DD + h * HD;
#pragma unroll
    for (int t = 0; t < 4; ++t)
      Ocat[rowbase + t * 16 + n] = f2bf(acc[t][r] * inv);
  }
}

// ---------- output projection: [B*S,D] @ Wp + bp -> f32 out ----------

__global__ __launch_bounds__(256) void k_proj(
    const bf16_t* __restrict__ Ocat, const bf16_t* __restrict__ WpT,
    const float* __restrict__ bp, float* __restrict__ out) {
  const int wg = blockIdx.x * 8 + (threadIdx.x >> 5);
  const int ng = wg & 15;                // 16 groups of 64 output cols
  const int mt = wg >> 4;                // 512 row tiles
  const int row0 = mt << 4;
  const int lane = threadIdx.x & 31;
  const int n = lane & 15, g = lane >> 4;

  v8f acc[4];
#pragma unroll
  for (int t = 0; t < 4; ++t) acc[t] = vzero8();

  for (int kk = 0; kk < DD; kk += 32) {
    v16bf a = load_a32(Ocat + (size_t)row0 * DD + kk, DD);
#pragma unroll
    for (int t = 0; t < 4; ++t) {
      v16bf bw = load_b32(WpT + (size_t)(ng * 64 + t * 16) * DD + kk, DD);
      acc[t] = wmma_bf16(a, bw, acc[t]);
    }
  }
#pragma unroll
  for (int t = 0; t < 4; ++t) {
    int col = ng * 64 + t * 16 + n;
    float bias = bp[col];
#pragma unroll
    for (int r = 0; r < 8; ++r) {
      int m = r + (g << 3);
      out[(size_t)(row0 + m) * DD + col] = acc[t][r] + bias;
    }
  }
}

// ---------- host launcher ----------

extern "C" void kernel_launch(void* const* d_in, const int* in_sizes, int n_in,
                              void* d_out, int out_size, void* d_ws, size_t ws_size,
                              hipStream_t stream) {
  const float* x  = (const float*)d_in[0];
  const float* Wq = (const float*)d_in[1];
  const float* bq = (const float*)d_in[2];
  const float* Wk = (const float*)d_in[3];
  const float* bk = (const float*)d_in[4];
  const float* Wv = (const float*)d_in[5];
  const float* bv = (const float*)d_in[6];
  const float* Wp = (const float*)d_in[7];
  const float* bp = (const float*)d_in[8];
  float* out = (float*)d_out;

  char* ws = (char*)d_ws;
  size_t off = 0;
  auto carve = [&](size_t bytes) -> char* {
    char* p = ws + off;
    off += (bytes + 255) & ~(size_t)255;
    return p;
  };
  bf16_t* xb  = (bf16_t*)carve((size_t)BB * SS * DD * 2);
  bf16_t* wqt = (bf16_t*)carve((size_t)HH * HD * DD * 2);
  bf16_t* wkt = (bf16_t*)carve((size_t)HH * HD * DD * 2);
  bf16_t* wvt = (bf16_t*)carve((size_t)HH * HD * DD * 2);
  bf16_t* wpt = (bf16_t*)carve((size_t)DD * DD * 2);
  bf16_t* Qb  = (bf16_t*)carve((size_t)BB * HH * SS * HD * 2);
  bf16_t* Kb  = (bf16_t*)carve((size_t)BB * HH * SS * HD * 2);
  bf16_t* VTb = (bf16_t*)carve((size_t)BB * HH * HD * SS * 2);
  bf16_t* Oc  = (bf16_t*)carve((size_t)BB * SS * DD * 2);

  const int nX = BB * SS * DD;
  k_cvt_x<<<(nX / 4 + 255) / 256, 256, 0, stream>>>(x, xb, nX);

  const int nW = HH * DD * HD;
  k_transpose<<<(nW + 255) / 256, 256, 0, stream>>>(Wq, wqt, HH, DD, HD);
  k_transpose<<<(nW + 255) / 256, 256, 0, stream>>>(Wk, wkt, HH, DD, HD);
  k_transpose<<<(nW + 255) / 256, 256, 0, stream>>>(Wv, wvt, HH, DD, HD);
  k_transpose<<<(DD * DD + 255) / 256, 256, 0, stream>>>(Wp, wpt, 1, DD, DD);

  k_qkv<<<1024, 256, 0, stream>>>(xb, wqt, wkt, wvt, bq, bk, bv, Qb, Kb, VTb);
  k_attn<<<1024, 256, 0, stream>>>(Qb, Kb, VTb, Oc);
  k_proj<<<1024, 256, 0, stream>>>(Oc, wpt, bp, out);
}
